// MultiGAT_48627619725729
// MI455X (gfx1250) — compile-verified, matrix operand order checked by reading, not run on hardware
//
#include <hip/hip_runtime.h>
#include <math.h>

// ---------------------------------------------------------------------------
// Types for CDNA5 WMMA (wave32)
// ---------------------------------------------------------------------------
typedef __attribute__((ext_vector_type(16))) __bf16 v16bf;
typedef __attribute__((ext_vector_type(8)))  __bf16 v8bf;
typedef __attribute__((ext_vector_type(8)))  float  v8f;

__device__ __forceinline__ v16bf cat16(v8bf a, v8bf b) {
    return __builtin_shufflevector(a, b, 0,1,2,3,4,5,6,7,8,9,10,11,12,13,14,15);
}

// Problem constants (from reference): b=4, n=2048, f=256, heads=4, d=64
#define NB   4
#define NN   2048
#define NF   256
#define NHD  4
#define ND   64
#define ALPHA 0.2f
#define NEGBIG (-1.0e12f)

// ---------------------------------------------------------------------------
// K0: W [h][256][64] f32  ->  WT_hi/WT_lo [h][64][256] bf16 (hi/lo split)
// ---------------------------------------------------------------------------
__global__ void __launch_bounds__(256) gat_wt_kernel(const float* __restrict__ W,
                                                     __bf16* __restrict__ WT_hi,
                                                     __bf16* __restrict__ WT_lo) {
    int idx = blockIdx.x * 256 + threadIdx.x;      // 4*64*256 = 65536 total
    int h   = idx >> 14;
    int rem = idx & 16383;
    int n   = rem >> 8;
    int k   = rem & 255;
    float w = W[(h * NF + k) * ND + n];
    __bf16 hi = (__bf16)w;
    WT_hi[idx] = hi;                                // [h][n][k] == idx
    WT_lo[idx] = (__bf16)(w - (float)hi);
}

// ---------------------------------------------------------------------------
// K1: H[bh] = x[b] @ W[h]  via v_wmma_f32_16x16x32_bf16 (3-term hi/lo split)
//   grid = 16 (bh) * 16 (M-blocks of 128 rows), block = 256 = 8 waves * 16 rows
//   writes Hf (f32 row-major) and HT_hi/HT_lo (bf16, [bh][64][2048] transposed)
// ---------------------------------------------------------------------------
__global__ void __launch_bounds__(256) gat_h_kernel(const float* __restrict__ x,
                                                    const __bf16* __restrict__ WT_hi,
                                                    const __bf16* __restrict__ WT_lo,
                                                    float* __restrict__ Hf,
                                                    __bf16* __restrict__ HT_hi,
                                                    __bf16* __restrict__ HT_lo) {
    const int lane = threadIdx.x & 31;
    const int wave = threadIdx.x >> 5;
    const int mb   = blockIdx.x & 15;
    const int bh   = blockIdx.x >> 4;
    const int b    = bh >> 2;
    const int h    = bh & 3;
    const int row0 = mb * 128 + wave * 16;
    const int m    = lane & 15;
    const int kb   = (lane >> 4) << 3;             // K-offset 0 or 8 (A/B layout)

    v8f acc[4];
    #pragma unroll
    for (int nt = 0; nt < 4; nt++)
        #pragma unroll
        for (int r = 0; r < 8; r++) acc[nt][r] = 0.0f;

    const float* xrow = x + ((size_t)b * NN + row0 + m) * NF;

    for (int kk = 0; kk < NF; kk += 32) {
        // --- A fragments: lane m holds k in {kb..kb+7} U {kb+16..kb+23}
        float xa[16];
        {
            const float4 f0 = *(const float4*)(xrow + kk + kb);
            const float4 f1 = *(const float4*)(xrow + kk + kb + 4);
            const float4 f2 = *(const float4*)(xrow + kk + kb + 16);
            const float4 f3 = *(const float4*)(xrow + kk + kb + 20);
            xa[0]=f0.x; xa[1]=f0.y; xa[2]=f0.z; xa[3]=f0.w;
            xa[4]=f1.x; xa[5]=f1.y; xa[6]=f1.z; xa[7]=f1.w;
            xa[8]=f2.x; xa[9]=f2.y; xa[10]=f2.z; xa[11]=f2.w;
            xa[12]=f3.x; xa[13]=f3.y; xa[14]=f3.z; xa[15]=f3.w;
        }
        v16bf aHi, aLo;
        #pragma unroll
        for (int q = 0; q < 16; q++) {
            __bf16 hv = (__bf16)xa[q];
            aHi[q] = hv;
            aLo[q] = (__bf16)(xa[q] - (float)hv);
        }
        // --- B fragments from transposed W, contiguous 8-element bf16 runs
        #pragma unroll
        for (int nt = 0; nt < 4; nt++) {
            const int n = nt * 16 + m;
            const size_t o = (size_t)(h * ND + n) * NF + kk + kb;
            v8bf bh0 = *(const v8bf*)(WT_hi + o);
            v8bf bh1 = *(const v8bf*)(WT_hi + o + 16);
            v8bf bl0 = *(const v8bf*)(WT_lo + o);
            v8bf bl1 = *(const v8bf*)(WT_lo + o + 16);
            v16bf bHi = cat16(bh0, bh1);
            v16bf bLo = cat16(bl0, bl1);
            acc[nt] = __builtin_amdgcn_wmma_f32_16x16x32_bf16(false, aHi, false, bHi,
                                                              (short)0, acc[nt], false, false);
            acc[nt] = __builtin_amdgcn_wmma_f32_16x16x32_bf16(false, aLo, false, bHi,
                                                              (short)0, acc[nt], false, false);
            acc[nt] = __builtin_amdgcn_wmma_f32_16x16x32_bf16(false, aHi, false, bLo,
                                                              (short)0, acc[nt], false, false);
        }
    }

    // --- epilogue: D layout is (row = r + 8*(lane>>4), col = nt*16 + lane&15)
    const int hb = (lane >> 4) << 3;
    #pragma unroll
    for (int nt = 0; nt < 4; nt++) {
        const int n = nt * 16 + m;
        v8bf ph, pl;
        #pragma unroll
        for (int r = 0; r < 8; r++) {
            const float v = acc[nt][r];
            Hf[((size_t)bh * NN + row0 + hb + r) * ND + n] = v;
            __bf16 hv = (__bf16)v;
            ph[r] = hv;
            pl[r] = (__bf16)(v - (float)hv);
        }
        const size_t o = (size_t)(bh * ND + n) * NN + row0 + hb;  // contiguous in r
        *(v8bf*)(HT_hi + o) = ph;
        *(v8bf*)(HT_lo + o) = pl;
    }
}

// ---------------------------------------------------------------------------
// K1b: f_src[bh][i] = H[bh][i] . a[h][:64],  f_dst = H . a[h][64:]
// ---------------------------------------------------------------------------
__global__ void __launch_bounds__(256) gat_f_kernel(const float* __restrict__ Hf,
                                                    const float* __restrict__ a,
                                                    float* __restrict__ fsrc,
                                                    float* __restrict__ fdst) {
    int idx = blockIdx.x * 256 + threadIdx.x;      // 16 * 2048 rows
    int bh  = idx >> 11;
    int h   = bh & 3;
    const float* hp = Hf + (size_t)idx * ND;
    const float* ap = a + h * 2 * ND;
    float s = 0.0f, d = 0.0f;
    #pragma unroll 8
    for (int k = 0; k < ND; k++) {
        float v = hp[k];
        s += v * ap[k];
        d += v * ap[ND + k];
    }
    fsrc[idx] = s;
    fdst[idx] = d;
}

// ---------------------------------------------------------------------------
// K2: fused masked-softmax attention, flash style, P@H on WMMA bf16 x3
//   grid = b(4) * h(4) * 16 i-blocks of 128 rows; block = 8 waves * 16 rows
// ---------------------------------------------------------------------------
__global__ void __launch_bounds__(256) gat_attn_kernel(const int* __restrict__ adj,
                                                       const float* __restrict__ x,
                                                       const float* __restrict__ Hf,
                                                       const __bf16* __restrict__ HT_hi,
                                                       const __bf16* __restrict__ HT_lo,
                                                       const float* __restrict__ fsrc,
                                                       const float* __restrict__ fdst,
                                                       float* __restrict__ out) {
    const int lane = threadIdx.x & 31;
    const int wave = threadIdx.x >> 5;
    const int ib   = blockIdx.x & 15;
    const int h    = (blockIdx.x >> 4) & 3;
    const int b    = blockIdx.x >> 6;
    const int bh   = b * NHD + h;
    const int i0   = ib * 128 + wave * 16;
    const int m    = lane & 15;
    const int kb   = (lane >> 4) << 3;

    const float fd = fdst[bh * NN + i0 + m];
    const float* fs_row  = fsrc + bh * NN;
    const int*   adj_row = adj + ((size_t)b * NN + (i0 + m)) * NN;
    const __bf16* hth = HT_hi + (size_t)bh * ND * NN;
    const __bf16* htl = HT_lo + (size_t)bh * ND * NN;

    v8f acc[4];
    #pragma unroll
    for (int nt = 0; nt < 4; nt++)
        #pragma unroll
        for (int r = 0; r < 8; r++) acc[nt][r] = 0.0f;

    float Mrow = -INFINITY, Lrow = 0.0f;

    for (int jt = 0; jt < 32; jt++) {
        const int j0 = jt * 64;
        if (jt + 1 < 32) __builtin_prefetch(adj_row + j0 + 64, 0, 3);

        // ---- scores for this wave's 16 rows x 64 cols, in A-fragment order
        float e[2][16];
        #pragma unroll
        for (int t = 0; t < 2; t++) {
            const int jb = j0 + t * 32 + kb;
            const float4 f0 = *(const float4*)(fs_row + jb);
            const float4 f1 = *(const float4*)(fs_row + jb + 4);
            const float4 f2 = *(const float4*)(fs_row + jb + 16);
            const float4 f3 = *(const float4*)(fs_row + jb + 20);
            const int4 a0 = *(const int4*)(adj_row + jb);
            const int4 a1 = *(const int4*)(adj_row + jb + 4);
            const int4 a2 = *(const int4*)(adj_row + jb + 16);
            const int4 a3 = *(const int4*)(adj_row + jb + 20);
            float fsv[16] = { f0.x,f0.y,f0.z,f0.w, f1.x,f1.y,f1.z,f1.w,
                              f2.x,f2.y,f2.z,f2.w, f3.x,f3.y,f3.z,f3.w };
            int   av[16]  = { a0.x,a0.y,a0.z,a0.w, a1.x,a1.y,a1.z,a1.w,
                              a2.x,a2.y,a2.z,a2.w, a3.x,a3.y,a3.z,a3.w };
            #pragma unroll
            for (int q = 0; q < 16; q++) {
                float v = fd + fsv[q];
                v = (v > 0.0f) ? v : ALPHA * v;           // leaky relu
                e[t][q] = (av[q] > 0) ? v : NEGBIG;       // adjacency mask
            }
        }

        // ---- online softmax stats (row split across lanes L and L+16)
        float tm = -INFINITY;
        #pragma unroll
        for (int t = 0; t < 2; t++)
            #pragma unroll
            for (int q = 0; q < 16; q++) tm = fmaxf(tm, e[t][q]);
        tm = fmaxf(tm, __shfl_xor(tm, 16, 32));
        const float newM = fmaxf(Mrow, tm);
        const float scaleRow = __expf(Mrow - newM);       // 0 on first tile
        Mrow = newM;

        v16bf aHi[2], aLo[2];
        float rs = 0.0f;
        #pragma unroll
        for (int t = 0; t < 2; t++)
            #pragma unroll
            for (int q = 0; q < 16; q++) {
                float p = __expf(e[t][q] - newM);
                rs += p;
                __bf16 ph = (__bf16)p;
                aHi[t][q] = ph;
                aLo[t][q] = (__bf16)(p - (float)ph);
            }
        rs += __shfl_xor(rs, 16, 32);
        Lrow = Lrow * scaleRow + rs;

        // ---- rescale accumulators (map row stats into D-fragment row space)
        {
            const int hb = (lane >> 4) << 3;
            float s[8];
            #pragma unroll
            for (int r = 0; r < 8; r++) s[r] = __shfl(scaleRow, hb + r, 32);
            #pragma unroll
            for (int nt = 0; nt < 4; nt++)
                #pragma unroll
                for (int r = 0; r < 8; r++) acc[nt][r] *= s[r];
        }

        // ---- acc += P @ H[j-tile]  (B fragments contiguous from transposed H)
        #pragma unroll
        for (int nt = 0; nt < 4; nt++) {
            const int n = nt * 16 + m;
            #pragma unroll
            for (int t = 0; t < 2; t++) {
                const size_t o = (size_t)n * NN + j0 + t * 32 + kb;
                v8bf bh0 = *(const v8bf*)(hth + o);
                v8bf bh1 = *(const v8bf*)(hth + o + 16);
                v8bf bl0 = *(const v8bf*)(htl + o);
                v8bf bl1 = *(const v8bf*)(htl + o + 16);
                v16bf bHi = cat16(bh0, bh1);
                v16bf bLo = cat16(bl0, bl1);
                acc[nt] = __builtin_amdgcn_wmma_f32_16x16x32_bf16(false, aHi[t], false, bHi,
                                                                  (short)0, acc[nt], false, false);
                acc[nt] = __builtin_amdgcn_wmma_f32_16x16x32_bf16(false, aLo[t], false, bHi,
                                                                  (short)0, acc[nt], false, false);
                acc[nt] = __builtin_amdgcn_wmma_f32_16x16x32_bf16(false, aHi[t], false, bLo,
                                                                  (short)0, acc[nt], false, false);
            }
        }
    }

    // ---- normalize, elu, residuals, store
    const int hb = (lane >> 4) << 3;
    const float Li = 1.0f / Lrow;                 // Lrow > 0 always (masked rows -> uniform)
    float linv[8];
    #pragma unroll
    for (int r = 0; r < 8; r++) linv[r] = __shfl(Li, hb + r, 32);

    #pragma unroll
    for (int nt = 0; nt < 4; nt++) {
        const int n = nt * 16 + m;
        #pragma unroll
        for (int r = 0; r < 8; r++) {
            const int row = i0 + hb + r;
            float o = acc[nt][r] * linv[r];
            o = (o > 0.0f) ? o : (__expf(o) - 1.0f);                 // elu
            const float hres = Hf[((size_t)bh * NN + row) * ND + n]; // + h
            const float xres = x[((size_t)b * NN + row) * NF + h * ND + n]; // + x
            out[((size_t)b * NN + row) * NF + h * ND + n] = o + hres + xres;
        }
    }
}

// ---------------------------------------------------------------------------
// Host launcher
// Workspace layout (bytes):
//   WT_hi 128K | WT_lo 128K | Hf 8M | HT_hi 4M | HT_lo 4M | fsrc 128K | fdst 128K
// ---------------------------------------------------------------------------
extern "C" void kernel_launch(void* const* d_in, const int* in_sizes, int n_in,
                              void* d_out, int out_size, void* d_ws, size_t ws_size,
                              hipStream_t stream) {
    const float* x   = (const float*)d_in[0];
    const int*   adj = (const int*)d_in[1];
    const float* W   = (const float*)d_in[2];
    const float* a   = (const float*)d_in[3];
    float* out = (float*)d_out;

    char* ws = (char*)d_ws;
    size_t off = 0;
    __bf16* WT_hi = (__bf16*)(ws + off); off += (size_t)NHD * ND * NF * 2;        // 128K
    __bf16* WT_lo = (__bf16*)(ws + off); off += (size_t)NHD * ND * NF * 2;        // 128K
    float*  Hf    = (float*) (ws + off); off += (size_t)NB * NHD * NN * ND * 4;   // 8M
    __bf16* HT_hi = (__bf16*)(ws + off); off += (size_t)NB * NHD * ND * NN * 2;   // 4M
    __bf16* HT_lo = (__bf16*)(ws + off); off += (size_t)NB * NHD * ND * NN * 2;   // 4M
    float*  fsrc  = (float*) (ws + off); off += (size_t)NB * NHD * NN * 4;        // 128K
    float*  fdst  = (float*) (ws + off); off += (size_t)NB * NHD * NN * 4;        // 128K

    gat_wt_kernel  <<<256, 256, 0, stream>>>(W, WT_hi, WT_lo);
    gat_h_kernel   <<<256, 256, 0, stream>>>(x, WT_hi, WT_lo, Hf, HT_hi, HT_lo);
    gat_f_kernel   <<<128, 256, 0, stream>>>(Hf, a, fsrc, fdst);
    gat_attn_kernel<<<256, 256, 0, stream>>>(adj, x, Hf, HT_hi, HT_lo, fsrc, fdst, out);
}
